// MultiHeadedAttention_56057913147529
// MI455X (gfx1250) — matrix operand order repaired and checked by reading, not compiled, not run on hardware
//
#include <hip/hip_runtime.h>
#include <hip/hip_bf16.h>

typedef __bf16 bf16_t;
typedef __attribute__((ext_vector_type(16))) __bf16 v16bf;
typedef __attribute__((ext_vector_type(8)))  __bf16 v8bf;
typedef __attribute__((ext_vector_type(4)))  __bf16 v4bf;
typedef __attribute__((ext_vector_type(8)))  float  v8f;

union V16 { v16bf v; v8bf h[2]; };

__device__ __forceinline__ v8f wmma_bf16(v16bf a, v16bf b, v8f c) {
    return __builtin_amdgcn_wmma_f32_16x16x32_bf16(
        /*neg_a=*/false, a, /*neg_b=*/false, b,
        /*c_mod=*/(short)0, c, /*reuse_a=*/false, /*reuse_b=*/false);
}

// LDS byte address = low 32 bits of the generic pointer (ISA 10.2: flat->LDS
// mapping truncates to addr[31:0]).
__device__ __forceinline__ unsigned lds_u32(const void* p) {
    return (unsigned)(size_t)p;
}

// Async 16B global -> LDS copy (ASYNCcnt tracked; per-lane addresses).
__device__ __forceinline__ void async_g2l_b128(unsigned lds_off,
                                               const void* gaddr) {
    asm volatile("global_load_async_to_lds_b128 %0, %1, off"
                 :: "v"(lds_off), "v"((unsigned long long)(size_t)gaddr)
                 : "memory");
}
__device__ __forceinline__ void wait_asynccnt0() {
    asm volatile("s_wait_asynccnt 0x0" ::: "memory");
}

// ---------------------------------------------------------------------------
// f32 -> bf16 conversion (vectorized x4)
// ---------------------------------------------------------------------------
__global__ void cvt_f32_to_bf16(const float* __restrict__ s,
                                bf16_t* __restrict__ d, int n) {
    int i = (blockIdx.x * blockDim.x + threadIdx.x) * 4;
    if (i < n) {
        float4 f = *(const float4*)(s + i);
        v4bf o;
        o[0] = (bf16_t)f.x; o[1] = (bf16_t)f.y;
        o[2] = (bf16_t)f.z; o[3] = (bf16_t)f.w;
        *(v4bf*)(d + i) = o;
    }
}

// ---------------------------------------------------------------------------
// Transpose-convert: W f32 [K,N] -> Wt bf16 [N,K]; 32x32 tiles via LDS so both
// global accesses are coalesced. One-time cost, removes all in-GEMM transposes.
// ---------------------------------------------------------------------------
__global__ __launch_bounds__(256)
void transpose_cvt(const float* __restrict__ W, bf16_t* __restrict__ Wt,
                   int K, int N) {
    __shared__ float t[32][33];
    const int n0 = blockIdx.x * 32, k0 = blockIdx.y * 32;
    const int tx = threadIdx.x, ty = threadIdx.y;   // 32 x 8
    #pragma unroll
    for (int i = 0; i < 4; ++i)
        t[ty * 4 + i][tx] = W[(size_t)(k0 + ty * 4 + i) * N + n0 + tx];
    __syncthreads();
    #pragma unroll
    for (int i = 0; i < 4; ++i)
        Wt[(size_t)(n0 + ty * 4 + i) * K + k0 + tx] = (bf16_t)t[tx][ty * 4 + i];
}

// ---------------------------------------------------------------------------
// bf16 WMMA GEMM: Out = A[M,K] @ Wt[N,K]^T  (128x128 tile, K-step 32,
// register->LDS ping-pong double buffering: next tile's global_load_b128s are
// issued before the WMMAs of the current tile; one barrier per K-step).
// mode 0: f32 out [M,N] + bias
// mode 1: bf16 out head-split [B,H,T,hd]
// mode 2: bf16 out head-split transposed [B,H,hd,T]
// ---------------------------------------------------------------------------
#define GLDS 40   // padded LDS row stride (32 K elems + 8)

__global__ __launch_bounds__(256)
void gemm_bf16_wmma(const bf16_t* __restrict__ A,
                    const bf16_t* __restrict__ Wt,
                    void* __restrict__ Out,
                    const float* __restrict__ bias,
                    int Kdim, int Ndim, int mode) {
    __shared__ bf16_t lA[2][128 * GLDS];     // [row][k]
    __shared__ bf16_t lB[2][128 * GLDS];     // [n][k]

    const int tid  = threadIdx.x;
    const int wave = tid >> 5, lane = tid & 31, ln = lane & 15;
    const bool hi  = lane >= 16;
    const int wm   = wave & 3;               // 4 waves over M (32 rows each)
    const int wn   = wave >> 2;              // 2 waves over N (64 cols each)
    const int m0   = blockIdx.x * 128;
    const int n0   = blockIdx.y * 128;

    // Staging assignment: thread -> (row, 16-elem column half)
    const int srow = tid >> 1, scc = (tid & 1) * 16;
    const bf16_t* gA = A  + (size_t)(m0 + srow) * Kdim + scc;
    const bf16_t* gB = Wt + (size_t)(n0 + srow) * Kdim + scc;

    // Prologue: stage K-tile 0
    v8bf ra0 = *(const v8bf*)gA, ra1 = *(const v8bf*)(gA + 8);
    v8bf rb0 = *(const v8bf*)gB, rb1 = *(const v8bf*)(gB + 8);
    *(v8bf*)(lA[0] + srow * GLDS + scc)     = ra0;
    *(v8bf*)(lA[0] + srow * GLDS + scc + 8) = ra1;
    *(v8bf*)(lB[0] + srow * GLDS + scc)     = rb0;
    *(v8bf*)(lB[0] + srow * GLDS + scc + 8) = rb1;
    __syncthreads();

    v8f acc[2][4] = {};
    const int nk = Kdim >> 5;

    for (int kt = 0; kt < nk; ++kt) {
        const int cur = kt & 1;
        const bool more = (kt + 1) < nk;
        if (more) {   // issue next tile's global loads early (overlap w/ WMMA)
            const bf16_t* pA = gA + (kt + 1) * 32;
            const bf16_t* pB = gB + (kt + 1) * 32;
            ra0 = *(const v8bf*)pA; ra1 = *(const v8bf*)(pA + 8);
            rb0 = *(const v8bf*)pB; rb1 = *(const v8bf*)(pB + 8);
        }

        V16 afr[2];
        #pragma unroll
        for (int mi = 0; mi < 2; ++mi) {
            const bf16_t* p = lA[cur] + (wm * 32 + mi * 16 + ln) * GLDS + (hi ? 8 : 0);
            afr[mi].h[0] = *(const v8bf*)p;
            afr[mi].h[1] = *(const v8bf*)(p + 16);
        }
        #pragma unroll
        for (int ni = 0; ni < 4; ++ni) {
            V16 bfr;
            const bf16_t* p = lB[cur] + (wn * 64 + ni * 16 + ln) * GLDS + (hi ? 8 : 0);
            bfr.h[0] = *(const v8bf*)p;
            bfr.h[1] = *(const v8bf*)(p + 16);
            #pragma unroll
            for (int mi = 0; mi < 2; ++mi)
                acc[mi][ni] = wmma_bf16(afr[mi].v, bfr.v, acc[mi][ni]);
        }

        if (more) {   // write next tile into the other buffer
            bf16_t* dA = lA[cur ^ 1] + srow * GLDS + scc;
            bf16_t* dB = lB[cur ^ 1] + srow * GLDS + scc;
            *(v8bf*)(dA)     = ra0;
            *(v8bf*)(dA + 8) = ra1;
            *(v8bf*)(dB)     = rb0;
            *(v8bf*)(dB + 8) = rb1;
        }
        __syncthreads();
    }

    // Epilogue (C layout: VGPR r -> row r (+8 for hi half), lane%16 -> col)
    #pragma unroll
    for (int mi = 0; mi < 2; ++mi)
    #pragma unroll
    for (int ni = 0; ni < 4; ++ni)
    #pragma unroll
    for (int r = 0; r < 8; ++r) {
        int row = m0 + wm * 32 + mi * 16 + r + (hi ? 8 : 0);
        int col = n0 + wn * 64 + ni * 16 + ln;
        float v = acc[mi][ni][r];
        if (mode == 0) {
            ((float*)Out)[(size_t)row * Ndim + col] = v + bias[col];
        } else {
            int b = row >> 11, t = row & 2047;     // T = 2048
            int h = col >> 6,  d = col & 63;       // hd = 64
            bf16_t* O = (bf16_t*)Out;
            if (mode == 1)
                O[(((size_t)(b * 16 + h) * 2048 + t) << 6) + d] = (bf16_t)v;
            else
                O[((size_t)(b * 16 + h) * 64 + d) * 2048 + t]   = (bf16_t)v;
        }
    }
}

// ---------------------------------------------------------------------------
// Flash attention, causal. Block = 128 Q rows x 1 head; 8 waves, wave w owns
// Q rows [w*16, w*16+16). KV tiles of 64, staged with DOUBLE-BUFFERED async
// global->LDS copies (ASYNCcnt): tile j+1's copies are issued right after the
// barrier and waited only after tile j's full compute, hiding the HBM/L2
// fetch under the WMMAs + softmax.
// Q,K in [B,H,T,64]; V in [B,H,64,T]. ctx out bf16 [B,T,1024].
// ---------------------------------------------------------------------------
__global__ __launch_bounds__(256)
void attn_flash_wmma(const bf16_t* __restrict__ Q,
                     const bf16_t* __restrict__ K,
                     const bf16_t* __restrict__ V,
                     bf16_t* __restrict__ ctx) {
    const int T = 2048, HD = 64;
    __shared__ bf16_t lK[2][64 * 72];  // [key][hd]
    __shared__ bf16_t lV[2][64 * 72];  // [hd][key]
    __shared__ bf16_t lP[128 * 72];    // [qrow][key] (per-wave private rows)

    const int tid  = threadIdx.x;
    const int wave = tid >> 5, lane = tid & 31, ln = lane & 15;
    const bool hi  = lane >= 16;
    const int bh   = blockIdx.y;              // b*H + h
    const int q0   = blockIdx.x * 128;

    const bf16_t* Qh = Q + (size_t)bh * T * HD;
    const bf16_t* Kh = K + (size_t)bh * T * HD;
    const bf16_t* Vh = V + (size_t)bh * HD * T;

    // Per-thread staging slots (16B x2 for K, x2 for V; per LDS buffer)
    const int srow = tid >> 2, scc = (tid & 3) * 16;
    const unsigned oK[2] = { lds_u32(lK[0] + srow * 72 + scc),
                             lds_u32(lK[1] + srow * 72 + scc) };
    const unsigned oV[2] = { lds_u32(lV[0] + srow * 72 + scc),
                             lds_u32(lV[1] + srow * 72 + scc) };

    // Issue one KV tile's async copies into LDS buffer `buf` (no wait).
    auto stage_kv = [&](int j, int buf) {
        const int kv0 = j * 64;
        const bf16_t* gk = Kh + (size_t)(kv0 + srow) * HD + scc;
        const bf16_t* gv = Vh + (size_t)srow * T + kv0 + scc;
        async_g2l_b128(oK[buf],      gk);
        async_g2l_b128(oK[buf] + 16, gk + 8);
        async_g2l_b128(oV[buf],      gv);
        async_g2l_b128(oV[buf] + 16, gv + 8);
    };

    // Q A-fragments (rows = wave's 16 q rows, K-dim = hd, 2 chunks of 32)
    V16 aq[2];
    {
        const bf16_t* base = Qh + (size_t)(q0 + wave * 16 + ln) * HD;
        #pragma unroll
        for (int f = 0; f < 2; ++f) {
            int c = f * 32 + (hi ? 8 : 0);
            aq[f].h[0] = *(const v8bf*)(base + c);
            aq[f].h[1] = *(const v8bf*)(base + c + 16);
        }
    }

    float mrow[8], lrow[8];
    v8f co[4] = {};
    #pragma unroll
    for (int r = 0; r < 8; ++r) { mrow[r] = -1e30f; lrow[r] = 0.f; }

    const int jmax = 2 * blockIdx.x + 1;
    // Prologue: stage tile 0 into buffer 0 and wait.
    stage_kv(0, 0);
    wait_asynccnt0();
    __syncthreads();

    for (int j = 0; j <= jmax; ++j) {
        const int kv0 = j * 64;
        const int cur = j & 1;
        // Kick off next tile's copies into the other buffer (safe: the
        // end-of-previous-iteration barrier proves no one still reads it).
        if (j < jmax) stage_kv(j + 1, cur ^ 1);

        // Scores: 4 key-subtiles of 16; K-dim=64 -> 2 chained WMMAs each
        v8f sc[4];
        #pragma unroll
        for (int ni = 0; ni < 4; ++ni) {
            V16 b0, b1;
            const bf16_t* p = lK[cur] + (ni * 16 + ln) * 72 + (hi ? 8 : 0);
            b0.h[0] = *(const v8bf*)p;        b0.h[1] = *(const v8bf*)(p + 16);
            b1.h[0] = *(const v8bf*)(p + 32); b1.h[1] = *(const v8bf*)(p + 48);
            v8f s = {};
            s = wmma_bf16(aq[0].v, b0.v, s);
            s = wmma_bf16(aq[1].v, b1.v, s);
            sc[ni] = s;
        }

        const bool needmask = (j >= 2 * (int)blockIdx.x);
        // Online softmax (row stats per C-layout row, half-wave reductions)
        #pragma unroll
        for (int r = 0; r < 8; ++r) {
            int qrow = q0 + wave * 16 + r + (hi ? 8 : 0);
            float mx = mrow[r];
            #pragma unroll
            for (int ni = 0; ni < 4; ++ni) {
                int key = kv0 + ni * 16 + ln;
                float v = sc[ni][r] * 0.125f;          // 1/sqrt(hd)
                if (needmask && key > qrow) v = -1e30f;
                sc[ni][r] = v;
                mx = fmaxf(mx, v);
            }
            #pragma unroll
            for (int m = 8; m; m >>= 1) mx = fmaxf(mx, __shfl_xor(mx, m, 32));
            float alpha = __expf(mrow[r] - mx);
            float psum = 0.f;
            #pragma unroll
            for (int ni = 0; ni < 4; ++ni) {
                float p = __expf(sc[ni][r] - mx);
                sc[ni][r] = p;
                psum += p;
            }
            #pragma unroll
            for (int m = 8; m; m >>= 1) psum += __shfl_xor(psum, m, 32);
            lrow[r] = lrow[r] * alpha + psum;
            mrow[r] = mx;
            #pragma unroll
            for (int nj = 0; nj < 4; ++nj) co[nj][r] *= alpha;
        }

        // P (C layout) -> LDS (wave-private rows), reload as A fragments
        #pragma unroll
        for (int r = 0; r < 8; ++r) {
            int prow = wave * 16 + r + (hi ? 8 : 0);
            #pragma unroll
            for (int ni = 0; ni < 4; ++ni)
                lP[prow * 72 + ni * 16 + ln] = (bf16_t)sc[ni][r];
        }
        // ctx += P(16x64) @ V(64x64): 2 key chunks x 4 hd subtiles
        #pragma unroll
        for (int kc = 0; kc < 2; ++kc) {
            V16 ap;
            const bf16_t* p = lP + (wave * 16 + ln) * 72 + kc * 32 + (hi ? 8 : 0);
            ap.h[0] = *(const v8bf*)p;
            ap.h[1] = *(const v8bf*)(p + 16);
            #pragma unroll
            for (int nj = 0; nj < 4; ++nj) {
                V16 bv;
                const bf16_t* q = lV[cur] + (nj * 16 + ln) * 72 + kc * 32 + (hi ? 8 : 0);
                bv.h[0] = *(const v8bf*)q;
                bv.h[1] = *(const v8bf*)(q + 16);
                co[nj] = wmma_bf16(ap.v, bv.v, co[nj]);
            }
        }
        // Close the tile: make sure our async copies for j+1 landed, then
        // rendezvous so everyone may read the freshly filled buffer.
        wait_asynccnt0();
        __syncthreads();
    }

    // Epilogue: ctx[b][t][h*64+d] = co / l
    const int h = bh & 15, b = bh >> 4;
    #pragma unroll
    for (int r = 0; r < 8; ++r) {
        float inv = 1.0f / lrow[r];
        int t = q0 + wave * 16 + r + (hi ? 8 : 0);
        #pragma unroll
        for (int nj = 0; nj < 4; ++nj) {
            int d = nj * 16 + ln;
            ctx[((size_t)(b * 2048 + t)) * 1024 + h * 64 + d] =
                (bf16_t)(co[nj][r] * inv);
        }
    }
}

// ---------------------------------------------------------------------------
extern "C" void kernel_launch(void* const* d_in, const int* in_sizes, int n_in,
                              void* d_out, int out_size, void* d_ws, size_t ws_size,
                              hipStream_t stream) {
    (void)in_sizes; (void)n_in; (void)out_size; (void)ws_size;
    const float* x  = (const float*)d_in[0];
    const float* Wq = (const float*)d_in[1];
    const float* Wk = (const float*)d_in[2];
    const float* Wv = (const float*)d_in[3];
    const float* Wo = (const float*)d_in[4];
    const float* bo = (const float*)d_in[5];

    const int B = 4, T = 2048, DIN = 1024, DOUT = 1024, H = 16;
    const size_t BT = (size_t)B * T;

    char* ws = (char*)d_ws;
    size_t off = 0;
    bf16_t* x_bf  = (bf16_t*)(ws + off); off += BT * DIN * 2;
    bf16_t* WqT   = (bf16_t*)(ws + off); off += (size_t)DIN * DOUT * 2;  // [N,K]
    bf16_t* WkT   = (bf16_t*)(ws + off); off += (size_t)DIN * DOUT * 2;
    bf16_t* WvT   = (bf16_t*)(ws + off); off += (size_t)DIN * DOUT * 2;
    bf16_t* WoT   = (bf16_t*)(ws + off); off += (size_t)DOUT * DOUT * 2;
    bf16_t* Qb    = (bf16_t*)(ws + off); off += BT * DOUT * 2;   // [B,H,T,64]
    bf16_t* Kb    = (bf16_t*)(ws + off); off += BT * DOUT * 2;   // [B,H,T,64]
    bf16_t* Vb    = (bf16_t*)(ws + off); off += BT * DOUT * 2;   // [B,H,64,T]
    bf16_t* Cb    = (bf16_t*)(ws + off); off += BT * DOUT * 2;   // [B,T,1024]

    // 1) converts (x straight; weights transposed to [N,K])
    {
        int nx = (int)(BT * DIN);
        cvt_f32_to_bf16<<<(nx / 4 + 255) / 256, 256, 0, stream>>>(x, x_bf, nx);
        dim3 tg(DOUT / 32, DIN / 32), tb(32, 8);
        transpose_cvt<<<tg, tb, 0, stream>>>(Wq, WqT, DIN, DOUT);
        transpose_cvt<<<tg, tb, 0, stream>>>(Wk, WkT, DIN, DOUT);
        transpose_cvt<<<tg, tb, 0, stream>>>(Wv, WvT, DIN, DOUT);
        transpose_cvt<<<tg, tb, 0, stream>>>(Wo, WoT, DOUT, DOUT);
    }

    // 2) QKV projections (64 x 8 tiles of 128x128)
    {
        dim3 g((unsigned)(BT / 128), DOUT / 128, 1);
        gemm_bf16_wmma<<<g, 256, 0, stream>>>(x_bf, WqT, Qb, nullptr, DIN, DOUT, 1);
        gemm_bf16_wmma<<<g, 256, 0, stream>>>(x_bf, WkT, Kb, nullptr, DIN, DOUT, 1);
        gemm_bf16_wmma<<<g, 256, 0, stream>>>(x_bf, WvT, Vb, nullptr, DIN, DOUT, 2);
    }

    // 3) causal flash attention
    {
        dim3 g(T / 128, B * H, 1);
        attn_flash_wmma<<<g, 256, 0, stream>>>(Qb, Kb, Vb, Cb);
    }

    // 4) output projection + bias, f32 out
    {
        dim3 g((unsigned)(BT / 128), DOUT / 128, 1);
        gemm_bf16_wmma<<<g, 256, 0, stream>>>(Cb, WoT, d_out, bo, DOUT, DOUT, 0);
    }
}
